// E_GCL_67327907332464
// MI455X (gfx1250) — matrix operand order repaired and checked by reading, not compiled, not run on hardware
//
#include <hip/hip_runtime.h>
#include <hip/hip_bf16.h>
#include <stdint.h>

typedef __attribute__((ext_vector_type(16))) __bf16 v16bf;
typedef __attribute__((ext_vector_type(8)))  float  v8f;

#define FDIM 128
#define NN   512
#define BB   2

// silu via hardware reciprocal (v_rcp_f32) instead of IEEE divide expansion
__device__ __forceinline__ float silu_f(float x) {
    return x * __builtin_amdgcn_rcpf(1.0f + __expf(-x));
}

__device__ __forceinline__ uint16_t f2bf(float x) {
    uint32_t u = __builtin_bit_cast(uint32_t, x);
    uint32_t r = (u + 0x7FFFu + ((u >> 16) & 1u)) >> 16;
    return (uint16_t)r;
}

// packed f32x2 -> bf16x2 (v_cvt_pk_bf16_f32 when available)
__device__ __forceinline__ uint32_t pack_bf16(float lo, float hi) {
#if __has_builtin(__builtin_amdgcn_cvt_pk_bf16_f32)
    auto v = __builtin_amdgcn_cvt_pk_bf16_f32(lo, hi);
    if constexpr (sizeof(v) == 4) {
        return __builtin_bit_cast(uint32_t, v);
    } else {
        return (uint32_t)f2bf(lo) | ((uint32_t)f2bf(hi) << 16);
    }
#else
    return (uint32_t)f2bf(lo) | ((uint32_t)f2bf(hi) << 16);
#endif
}

struct Frag32 { uint4 lo, hi; };

__device__ __forceinline__ v16bf load_frag(const uint16_t* p) {
    Frag32 f;
    const uint4* q = (const uint4*)p;
    f.lo = q[0];
    f.hi = q[1];
    return __builtin_bit_cast(v16bf, f);
}

// ---------------------------------------------------------------------------
// Precompute pre_src[b,j,f] = (h_j*mask)@W1[:F] and pre_tgt[b,i,f] = (h_i*mask)@W1[F:2F] + b1
// ---------------------------------------------------------------------------
__global__ __launch_bounds__(128) void pre_kernel(
    const float* __restrict__ h, const float* __restrict__ mask,
    const float* __restrict__ w1, const float* __restrict__ b1,
    float* __restrict__ pre_src, float* __restrict__ pre_tgt)
{
    int bi = blockIdx.x;         // 0..B*N-1
    int f  = threadIdx.x;        // 0..127
    __shared__ float hm[FDIM];
    float mk = mask[bi];
    hm[f] = h[bi * FDIM + f] * mk;
    __syncthreads();
    float s1 = 0.f, s2 = 0.f;
    #pragma unroll 8
    for (int k = 0; k < FDIM; ++k) {
        float hv = hm[k];
        s1 += hv * w1[k * FDIM + f];
        s2 += hv * w1[(FDIM + k) * FDIM + f];
    }
    pre_src[bi * FDIM + f] = s1;
    pre_tgt[bi * FDIM + f] = s2 + b1[f];
}

// ---------------------------------------------------------------------------
// Pack edge_w2 / coord_w1 into bf16 WMMA B-fragment order:
// element e of lane L for tile (t,c): k = c*32 + (L/16)*16 + e ; n = t*16 + (L%16)
// ---------------------------------------------------------------------------
__global__ __launch_bounds__(256) void pack_kernel(
    const float* __restrict__ w2, const float* __restrict__ cw1,
    uint16_t* __restrict__ w2f, uint16_t* __restrict__ cw1f)
{
    int idx = blockIdx.x * 256 + threadIdx.x;    // 0..32767
    const float* W = (idx < 16384) ? w2 : cw1;
    uint16_t*    D = (idx < 16384) ? w2f : cw1f;
    int id   = idx & 16383;
    int e    = id & 15;
    int lane = (id >> 4) & 31;
    int c    = (id >> 9) & 3;
    int t    = id >> 11;
    int k = c * 32 + (lane >> 4) * 16 + e;
    int n = t * 16 + (lane & 15);
    D[id] = f2bf(W[k * FDIM + n]);
}

// ---------------------------------------------------------------------------
// Fused edge/coord kernel. One wave per target node i; loops j in tiles of 16.
// Per tile: hidden1 (factorized layer1) -> WMMA GEMM (edge_w2) -> m
//           -> WMMA GEMM (coord_w1) -> embed (coord_w2 dot) -> trans/agg accum.
// ---------------------------------------------------------------------------
__global__ __launch_bounds__(256) void edge_kernel(
    const float* __restrict__ coord, const float* __restrict__ mask,
    const float* __restrict__ w1,                    // edge_w1, row 2F = radial weights
    const float* __restrict__ b2,                    // edge_b2
    const float* __restrict__ cb1,                   // coord_b1
    const float* __restrict__ cw2,                   // coord_w2 [F,1]
    const float* __restrict__ pre_src, const float* __restrict__ pre_tgt,
    const uint16_t* __restrict__ w2f_in, const uint16_t* __restrict__ cw1f_in,
    float* __restrict__ agg, float* __restrict__ coord_out)
{
    const int lane = threadIdx.x & 31;
    const int wave = threadIdx.x >> 5;
    const int gi   = blockIdx.x * 8 + wave;          // 0..B*N-1
    const int b    = gi / NN;
    const int col  = lane & 15;
    const int half = lane >> 4;

    const int S = 136;                                // bf16 row stride (272B, 16B-mult)
    __shared__ __align__(16) uint16_t h1buf[8][16 * S];
    __shared__ __align__(16) float    scbuf[8][48];   // [0:16) radial [16:32) mask2 [32:48) embed
    uint16_t* h1    = h1buf[wave];
    float*    rad_l = scbuf[wave];
    float*    m2_l  = scbuf[wave] + 16;
    float*    emb_l = scbuf[wave] + 32;

    // mutable copies so an empty asm can make them "loop-variant" each iteration
    const uint16_t* w2f  = w2f_in;
    const uint16_t* cw1f = cw1f_in;

    // per-lane constants
    const int f0 = lane * 4;
    float4 pt = *(const float4*)(pre_tgt + gi * FDIM + f0);
    float4 wr = *(const float4*)(w1 + 2 * FDIM * FDIM + f0);   // radial row of edge_w1
    float b2c[8], cb1c[8], cw2c[8];
    #pragma unroll
    for (int t = 0; t < 8; ++t) {
        b2c[t]  = b2 [t * 16 + col];
        cb1c[t] = cb1[t * 16 + col];
        cw2c[t] = cw2[t * 16 + col];
    }

    float mi  = mask[gi];
    float cix = coord[gi * 3 + 0] * mi;
    float ciy = coord[gi * 3 + 1] * mi;
    float ciz = coord[gi * 3 + 2] * mi;

    float agg_acc[8] = {0.f, 0.f, 0.f, 0.f, 0.f, 0.f, 0.f, 0.f};
    float tx = 0.f, ty = 0.f, tz = 0.f, dsum = 0.f;

    for (int j0 = 0; j0 < NN; j0 += 16) {
        // Defeat LICM on the weight-fragment loads: keeping them inside the loop
        // avoids 2KB/lane of hoisted operands spilling to scratch; the reloads hit
        // WGP$ and are shared by all waves of the block.
        asm volatile("" : "+s"(w2f), "+s"(cw1f));

        // ---- geometry: coord_diff, radial, mask2 (col-th edge; both halves mirror)
        int   j   = b * NN + j0 + col;
        float mj  = mask[j];
        float m2  = mi * mj;
        float cdx = cix - coord[j * 3 + 0] * mj;
        float cdy = ciy - coord[j * 3 + 1] * mj;
        float cdz = ciz - coord[j * 3 + 2] * mj;
        float radv = (cdx * cdx + cdy * cdy + cdz * cdz) * m2;
        cdx *= m2; cdy *= m2; cdz *= m2;
        if (lane < 16) { rad_l[lane] = radv; m2_l[lane] = m2; }
        dsum += m2;

        if (j0 + 16 < NN)
            __builtin_prefetch(pre_src + (b * NN + j0 + 16) * FDIM + f0, 0, 1);

        // ---- hidden1 = silu(pre_src[j] + pre_tgt[i] + radial*w1_rad) -> bf16 LDS tile
        #pragma unroll
        for (int e = 0; e < 16; ++e) {
            float4 ps = *(const float4*)(pre_src + (b * NN + j0 + e) * FDIM + f0);
            float  r  = rad_l[e];
            float v0 = silu_f(ps.x + pt.x + r * wr.x);
            float v1 = silu_f(ps.y + pt.y + r * wr.y);
            float v2 = silu_f(ps.z + pt.z + r * wr.z);
            float v3 = silu_f(ps.w + pt.w + r * wr.w);
            uint2 pk;
            pk.x = pack_bf16(v0, v1);
            pk.y = pack_bf16(v2, v3);
            *(uint2*)(h1 + e * S + f0) = pk;
        }

        // ---- A fragments (16x128 as 4 chunks of 16x32, contiguous k = 16*half + s, as packed B)
        v16bf a[4];
        #pragma unroll
        for (int c = 0; c < 4; ++c) {
            int koff = c * 32 + half * 16;
            Frag32 fr;
            fr.lo = *(const uint4*)(h1 + col * S + koff);
            fr.hi = *(const uint4*)(h1 + col * S + koff + 8);
            a[c] = __builtin_bit_cast(v16bf, fr);
        }

        // per-row mask2 for this lane's 8 C-rows
        float m2r[8];
        {
            float4 x = *(const float4*)(m2_l + half * 8);
            float4 y = *(const float4*)(m2_l + half * 8 + 4);
            m2r[0]=x.x; m2r[1]=x.y; m2r[2]=x.z; m2r[3]=x.w;
            m2r[4]=y.x; m2r[5]=y.y; m2r[6]=y.z; m2r[7]=y.w;
        }

        // ---- GEMM 1: m = silu(hidden1 @ edge_w2 + b2) * mask2 ; accumulate agg; m -> h1 (bf16)
        #pragma unroll
        for (int t = 0; t < 8; ++t) {
            v8f acc = {0.f, 0.f, 0.f, 0.f, 0.f, 0.f, 0.f, 0.f};
            #pragma unroll
            for (int c = 0; c < 4; ++c) {
                v16bf bb = load_frag(w2f + ((t * 4 + c) * 32 + lane) * 16);
                acc = __builtin_amdgcn_wmma_f32_16x16x32_bf16(
                        false, a[c], false, bb, (short)0, acc, false, false);
            }
            float s = 0.f;
            float mv[8];
            #pragma unroll
            for (int r = 0; r < 8; ++r) {
                mv[r] = silu_f(acc[r] + b2c[t]) * m2r[r];
                s += mv[r];
            }
            #pragma unroll
            for (int r = 0; r < 8; ++r)
                h1[(half * 8 + r) * S + t * 16 + col] = f2bf(mv[r]);
            s += __shfl_xor(s, 16, 32);
            agg_acc[t] += s;
        }

        // ---- A fragments of m
        v16bf a2[4];
        #pragma unroll
        for (int c = 0; c < 4; ++c) {
            int koff = c * 32 + half * 16;
            Frag32 fr;
            fr.lo = *(const uint4*)(h1 + col * S + koff);
            fr.hi = *(const uint4*)(h1 + col * S + koff + 8);
            a2[c] = __builtin_bit_cast(v16bf, fr);
        }

        // ---- GEMM 2: p = silu(m @ coord_w1 + cb1); embed = p @ coord_w2
        float eacc[8] = {0.f, 0.f, 0.f, 0.f, 0.f, 0.f, 0.f, 0.f};
        #pragma unroll
        for (int t = 0; t < 8; ++t) {
            v8f acc = {0.f, 0.f, 0.f, 0.f, 0.f, 0.f, 0.f, 0.f};
            #pragma unroll
            for (int c = 0; c < 4; ++c) {
                v16bf bb = load_frag(cw1f + ((t * 4 + c) * 32 + lane) * 16);
                acc = __builtin_amdgcn_wmma_f32_16x16x32_bf16(
                        false, a2[c], false, bb, (short)0, acc, false, false);
            }
            #pragma unroll
            for (int r = 0; r < 8; ++r)
                eacc[r] += silu_f(acc[r] + cb1c[t]) * cw2c[t];
        }
        // reduce over 16 columns (stays within each 16-lane half)
        #pragma unroll
        for (int m = 1; m < 16; m <<= 1)
            #pragma unroll
            for (int r = 0; r < 8; ++r)
                eacc[r] += __shfl_xor(eacc[r], m, 32);
        if (col == 0)
            #pragma unroll
            for (int r = 0; r < 8; ++r)
                emb_l[half * 8 + r] = eacc[r];

        // ---- trans accumulation (edge index = col; halves duplicate, reduced later)
        float emb = emb_l[col];
        tx += cdx * emb;
        ty += cdy * emb;
        tz += cdz * emb;
    }

    // ---- final reductions over lanes 0..15
    #pragma unroll
    for (int m = 1; m < 16; m <<= 1) {
        dsum += __shfl_xor(dsum, m, 32);
        tx   += __shfl_xor(tx,   m, 32);
        ty   += __shfl_xor(ty,   m, 32);
        tz   += __shfl_xor(tz,   m, 32);
    }
    if (lane == 0) {
        float inv = __builtin_amdgcn_rcpf(dsum + 1e-10f);
        coord_out[gi * 3 + 0] = (cix + tx * inv) * mi;
        coord_out[gi * 3 + 1] = (ciy + ty * inv) * mi;
        coord_out[gi * 3 + 2] = (ciz + tz * inv) * mi;
    }
    if (lane < 16) {
        #pragma unroll
        for (int t = 0; t < 8; ++t)
            agg[gi * FDIM + t * 16 + lane] = agg_acc[t];
    }
}

// ---------------------------------------------------------------------------
// Node MLP: h_new = (h + silu([h,agg]@nw1 + nb1)@nw2 + nb2) * mask
// ---------------------------------------------------------------------------
__global__ __launch_bounds__(128) void node_kernel(
    const float* __restrict__ h, const float* __restrict__ mask,
    const float* __restrict__ agg,
    const float* __restrict__ nw1, const float* __restrict__ nb1,
    const float* __restrict__ nw2, const float* __restrict__ nb2,
    float* __restrict__ hout)
{
    int gi = blockIdx.x;
    int f  = threadIdx.x;
    __shared__ float cat[2 * FDIM];
    __shared__ float hid[FDIM];
    float mk = mask[gi];
    float hv = h[gi * FDIM + f] * mk;
    cat[f]        = hv;                         // h already masked
    cat[FDIM + f] = agg[gi * FDIM + f] * mk;
    __syncthreads();
    float s = nb1[f];
    #pragma unroll 8
    for (int k = 0; k < 2 * FDIM; ++k)
        s += cat[k] * nw1[k * FDIM + f];
    hid[f] = silu_f(s);
    __syncthreads();
    float o = nb2[f];
    #pragma unroll 8
    for (int k = 0; k < FDIM; ++k)
        o += hid[k] * nw2[k * FDIM + f];
    hout[gi * FDIM + f] = (hv + o) * mk;
}

// ---------------------------------------------------------------------------
extern "C" void kernel_launch(void* const* d_in, const int* in_sizes, int n_in,
                              void* d_out, int out_size, void* d_ws, size_t ws_size,
                              hipStream_t stream)
{
    const float* h    = (const float*)d_in[0];
    const float* crd  = (const float*)d_in[1];
    const float* mask = (const float*)d_in[2];
    const float* ew1  = (const float*)d_in[3];
    const float* eb1  = (const float*)d_in[4];
    const float* ew2  = (const float*)d_in[5];
    const float* eb2  = (const float*)d_in[6];
    const float* nw1  = (const float*)d_in[7];
    const float* nb1  = (const float*)d_in[8];
    const float* nw2  = (const float*)d_in[9];
    const float* nb2  = (const float*)d_in[10];
    const float* cw1  = (const float*)d_in[11];
    const float* cb1  = (const float*)d_in[12];
    const float* cw2  = (const float*)d_in[13];

    const int BNF = BB * NN * FDIM;              // 131072
    float* ws       = (float*)d_ws;
    float* pre_src  = ws;
    float* pre_tgt  = ws + BNF;
    float* agg      = ws + 2 * BNF;
    uint16_t* w2f   = (uint16_t*)(ws + 3 * BNF);
    uint16_t* cw1f  = w2f + FDIM * FDIM;

    float* hout = (float*)d_out;
    float* cout = hout + BNF;

    pre_kernel <<<BB * NN, 128, 0, stream>>>(h, mask, ew1, eb1, pre_src, pre_tgt);
    pack_kernel<<<128, 256, 0, stream>>>(ew2, cw1, w2f, cw1f);
    edge_kernel<<<BB * NN / 8, 256, 0, stream>>>(crd, mask, ew1, eb2, cb1, cw2,
                                                 pre_src, pre_tgt, w2f, cw1f,
                                                 agg, cout);
    node_kernel<<<BB * NN, 128, 0, stream>>>(h, mask, agg, nw1, nb1, nw2, nb2, hout);
}
